// MoEGate_13597866459200
// MI455X (gfx1250) — compile-verified, matrix operand order checked by reading, not run on hardware
//
#include <hip/hip_runtime.h>
#include <hip/hip_bf16.h>

// ---------------------------------------------------------------------------
// MoE gate for MI455X (gfx1250):
//   pass 0: gate_w f32 -> bf16 in workspace (2 MB, L2-resident)
//   pass 1: logits = H @ W^T via v_wmma_f32_16x16x32_bf16 (f32 accumulate)
//           block = 4 waves (2M x 2N), wave tile 32x64 -> 8 WMMA / k-step
//           grid (128, 2): hidden streamed once from HBM, W + 2nd N-pass
//           served from the 192 MB L2. HBM floor ~146 MB ≈ 6.3 us @ 23.3 TB/s.
//   pass 2: sigmoid + group-limited top-2 + normalize + 2.5x scaling
// ---------------------------------------------------------------------------

#define N_TOKENS   8192
#define N_EXPERTS  256
#define D_MODEL    4096
#define N_GROUP    4
#define GROUP_SZ   (N_EXPERTS / N_GROUP)   // 64
#define TOPK_GROUP 2
#define ROUTED_SCALING 2.5f
#define EPS        1e-10f

typedef __attribute__((ext_vector_type(16))) __bf16 v16bf;
typedef __attribute__((ext_vector_type(8)))  __bf16 v8bf;
typedef __attribute__((ext_vector_type(4)))  __bf16 v4bf;
typedef __attribute__((ext_vector_type(8)))  float  v8f;

// ---------------------------------------------------------------------------
// Pass 0: convert gate weights to bf16 (1M elements, vectorized x4)
// ---------------------------------------------------------------------------
__global__ __launch_bounds__(256) void convert_w_kernel(const float* __restrict__ w,
                                                        __bf16* __restrict__ wbf) {
    int i = blockIdx.x * blockDim.x + threadIdx.x;           // 0 .. 262143
    float4 v = ((const float4*)w)[i];
    v4bf o;
    o[0] = (__bf16)v.x; o[1] = (__bf16)v.y; o[2] = (__bf16)v.z; o[3] = (__bf16)v.w;
    ((v4bf*)wbf)[i] = o;
}

// ---------------------------------------------------------------------------
// Pass 1: GEMM.
// A fragment (16x32 bf16): lane = M row, per-lane K = two contiguous runs of 8
//   starting at k0 + 8*(lane>=16) and k0 + 16 + 8*(lane>=16).
// B fragment (32x16 bf16): identical packing with lane = N (gate_w is [E,D]).
// ---------------------------------------------------------------------------
__device__ __forceinline__ v16bf load_a_frag(const float* __restrict__ row, int k0, int g) {
    const float4* p0 = (const float4*)(row + k0 + 8 * g);        // K run 0 (8 floats)
    const float4* p1 = (const float4*)(row + k0 + 16 + 8 * g);   // K run 1 (8 floats)
    float4 x0 = p0[0], x1 = p0[1];
    float4 y0 = p1[0], y1 = p1[1];
    union { v16bf v; __bf16 e[16]; } a;
    a.e[0]  = (__bf16)x0.x; a.e[1]  = (__bf16)x0.y; a.e[2]  = (__bf16)x0.z; a.e[3]  = (__bf16)x0.w;
    a.e[4]  = (__bf16)x1.x; a.e[5]  = (__bf16)x1.y; a.e[6]  = (__bf16)x1.z; a.e[7]  = (__bf16)x1.w;
    a.e[8]  = (__bf16)y0.x; a.e[9]  = (__bf16)y0.y; a.e[10] = (__bf16)y0.z; a.e[11] = (__bf16)y0.w;
    a.e[12] = (__bf16)y1.x; a.e[13] = (__bf16)y1.y; a.e[14] = (__bf16)y1.z; a.e[15] = (__bf16)y1.w;
    return a.v;
}

__device__ __forceinline__ v16bf load_b_frag(const __bf16* __restrict__ row, int k0, int g) {
    v8bf lo = *(const v8bf*)(row + k0 + 8 * g);
    v8bf hi = *(const v8bf*)(row + k0 + 16 + 8 * g);
    return __builtin_shufflevector(lo, hi, 0, 1, 2, 3, 4, 5, 6, 7,
                                           8, 9, 10, 11, 12, 13, 14, 15);
}

__device__ __forceinline__ void store_c_tile(float* __restrict__ logits, v8f acc,
                                             int rowBase, int colBase, int g, int mr) {
#pragma unroll
    for (int r = 0; r < 8; ++r) {
        logits[(size_t)(rowBase + 8 * g + r) * N_EXPERTS + colBase + mr] = acc[r];
    }
}

// Block: 128 threads = 4 waves as 2(M) x 2(N). Wave tile: 32 rows x 64 cols.
// Block tile: 64 tokens x 128 experts. Grid: (8192/64, 256/128) = (128, 2).
__global__ __launch_bounds__(128) void gemm_logits_kernel(const float* __restrict__ hidden,
                                                          const __bf16* __restrict__ wbf,
                                                          float* __restrict__ logits) {
    const int tid   = threadIdx.x;
    const int lane  = tid & 31;
    const int wid   = tid >> 5;
    const int waveM = wid & 1;                 // 0..1
    const int waveN = wid >> 1;                // 0..1
    const int m0    = blockIdx.x * 64  + waveM * 32;
    const int n0    = blockIdx.y * 128 + waveN * 64;

    const int g  = lane >> 4;                  // half-wave selector
    const int mr = lane & 15;                  // row/col within fragment

    const float* arow0 = hidden + (size_t)(m0 + mr)      * D_MODEL;
    const float* arow1 = hidden + (size_t)(m0 + 16 + mr) * D_MODEL;
    const __bf16* brow[4];
#pragma unroll
    for (int j = 0; j < 4; ++j)
        brow[j] = wbf + (size_t)(n0 + 16 * j + mr) * D_MODEL;

    v8f acc0[4], acc1[4];
#pragma unroll
    for (int j = 0; j < 4; ++j) {
        acc0[j] = (v8f){0.f,0.f,0.f,0.f,0.f,0.f,0.f,0.f};
        acc1[j] = (v8f){0.f,0.f,0.f,0.f,0.f,0.f,0.f,0.f};
    }

    for (int k0 = 0; k0 < D_MODEL; k0 += 32) {
        // keep the A stream ahead of us (emits global_prefetch_b8)
        __builtin_prefetch(arow0 + k0 + 128, 0, 1);
        __builtin_prefetch(arow1 + k0 + 128, 0, 1);

        v16bf a0 = load_a_frag(arow0, k0, g);
        v16bf a1 = load_a_frag(arow1, k0, g);

#pragma unroll
        for (int j = 0; j < 4; ++j) {
            v16bf b = load_b_frag(brow[j], k0, g);
            acc0[j] = __builtin_amdgcn_wmma_f32_16x16x32_bf16(false, a0, false, b,
                                                              (short)0, acc0[j], false, false);
            acc1[j] = __builtin_amdgcn_wmma_f32_16x16x32_bf16(false, a1, false, b,
                                                              (short)0, acc1[j], false, false);
        }
    }

#pragma unroll
    for (int j = 0; j < 4; ++j) {
        store_c_tile(logits, acc0[j], m0,      n0 + 16 * j, g, mr);
        store_c_tile(logits, acc1[j], m0 + 16, n0 + 16 * j, g, mr);
    }
}

// ---------------------------------------------------------------------------
// Pass 2: routing. One thread per token: sigmoid, top-2 per 64-expert group
// (strict > keeps the lowest index on ties, matching jax.lax.top_k),
// normalize by (sum + eps), scale by 2.5. Indices stored value-converted.
// ---------------------------------------------------------------------------
__global__ __launch_bounds__(256) void route_kernel(const float* __restrict__ logits,
                                                    float* __restrict__ out_idx,
                                                    float* __restrict__ out_w) {
    int t = blockIdx.x * blockDim.x + threadIdx.x;
    if (t >= N_TOKENS) return;
    const float* row = logits + (size_t)t * N_EXPERTS;

    float wv[N_GROUP * TOPK_GROUP];
    int   wi[N_GROUP * TOPK_GROUP];
    float wsum = 0.0f;

#pragma unroll
    for (int grp = 0; grp < N_GROUP; ++grp) {
        float best1 = -1e30f, best2 = -1e30f;
        int   i1 = 0, i2 = 0;
        for (int j = 0; j < GROUP_SZ; ++j) {
            float s = 1.0f / (1.0f + expf(-row[grp * GROUP_SZ + j]));
            if (s > best1)      { best2 = best1; i2 = i1; best1 = s; i1 = j; }
            else if (s > best2) { best2 = s; i2 = j; }
        }
        wv[grp * 2]     = best1; wi[grp * 2]     = grp * GROUP_SZ + i1;
        wv[grp * 2 + 1] = best2; wi[grp * 2 + 1] = grp * GROUP_SZ + i2;
        wsum += best1 + best2;
    }

    float scale = ROUTED_SCALING / (wsum + EPS);
#pragma unroll
    for (int k = 0; k < N_GROUP * TOPK_GROUP; ++k) {
        out_idx[(size_t)t * 8 + k] = (float)wi[k];
        out_w[(size_t)t * 8 + k]   = wv[k] * scale;
    }
}

// ---------------------------------------------------------------------------
extern "C" void kernel_launch(void* const* d_in, const int* in_sizes, int n_in,
                              void* d_out, int out_size, void* d_ws, size_t ws_size,
                              hipStream_t stream) {
    const float* hidden = (const float*)d_in[0];   // [8192, 4096] f32
    const float* gate_w = (const float*)d_in[1];   // [256, 4096]  f32

    float* out      = (float*)d_out;
    float* out_idx  = out;                               // 8192*8
    float* out_wt   = out + (size_t)N_TOKENS * 8;        // 8192*8
    float* logits   = out + (size_t)N_TOKENS * 8 * 2;    // 8192*256

    __bf16* wbf = (__bf16*)d_ws;                         // 2 MB bf16 copy of gate_w

    // pass 0: 1,048,576 elems / 4 per thread / 256 per block = 1024 blocks
    convert_w_kernel<<<1024, 256, 0, stream>>>(gate_w, wbf);

    // pass 1: block tile 64 tokens x 128 experts
    dim3 grid(N_TOKENS / 64, N_EXPERTS / 128);
    gemm_logits_kernel<<<grid, 128, 0, stream>>>(hidden, wbf, logits);

    // pass 2: one thread per token
    route_kernel<<<N_TOKENS / 256, 256, 0, stream>>>(logits, out_idx, out_wt);
}
